// DeepVCP_24257975288100
// MI455X (gfx1250) — matrix operand order repaired and checked by reading, not compiled, not run on hardware
//
#include <hip/hip_runtime.h>
#include <hip/hip_bf16.h>
#include <math.h>

typedef __attribute__((ext_vector_type(16))) _Float16 v16h;
typedef __attribute__((ext_vector_type(8)))  float    v8f;

#define NPTS 65536
#define KKEY 64

// ---------------------------------------------------------------------------
// WMMA wrapper: D = A(16x32 f16) * B(32x16 f16) + C(16x16 f32), wave32.
// ---------------------------------------------------------------------------
__device__ inline v8f wmma16(v16h a, v16h b, v8f c) {
  return __builtin_amdgcn_wmma_f32_16x16x32_f16(
      /*neg_a=*/false, a, /*neg_b=*/false, b,
      /*c_mod=*/(short)0, c, /*reuse_a=*/false, /*reuse_b=*/false);
}

// Build a B operand (32 K x 16 cols) from an LDS activation tile act[K][16].
// Branch-free: clamped LDS reads + selects; bound checks constant-fold when
// the whole K window is in range (layers 2 and a).
// Contiguous K order (A staged the same): lanes 0-15 hold column N=lane with K = 0..15 of the window,
// lanes 16-31 hold the same columns with K = 16..31.
__device__ inline v16h load_B_lds(const float* act, int kbase, int Kin, int lane) {
  const int half_ = lane >> 4;
  const int lcol  = lane & 15;
  v16h b;
#pragma unroll
  for (int v = 0; v < 8; ++v) {
    const int  k   = kbase + half_ * 16 + 2 * v;
    const bool ok0 = (k     < Kin);
    const bool ok1 = (k + 1 < Kin);
    float x0 = act[(ok0 ? k     : 0) * 16 + lcol];
    float x1 = act[(ok1 ? k + 1 : 0) * 16 + lcol];
    b[2 * v]     = (_Float16)(ok0 ? x0 : 0.f);
    b[2 * v + 1] = (_Float16)(ok1 ? x1 : 0.f);
  }
  return b;
}

// Store a D accumulator (16x16 f32) to the LDS activation tile, bias + ReLU.
// D layout: VGPR r holds row M = mbase + r + 8*(lane>=16), col N = lane&15.
__device__ inline void store_D_relu(float* act, v8f acc, const float* bias_lds,
                                    int mbase, int Rout, int lane) {
  const int half_ = lane >> 4;
  const int lcol  = lane & 15;
#pragma unroll
  for (int r = 0; r < 8; ++r) {
    const int m = mbase + r + half_ * 8;
    if (m < Rout) {                       // folds to true for full tiles
      float val = acc[r] + bias_lds[m];
      act[m * 16 + lcol] = fmaxf(val, 0.f);
    }
  }
}

// ---------------------------------------------------------------------------
// Kernel 1: fused per-point score MLP for batch 0.
// One wave = one 16-point tile; 8 waves / block.
// All 9 WMMA A-tiles are staged once per block into LDS in the exact
// per-lane operand layout (lane-contiguous 32B), so A loads are 2x b128.
//   6 ->32 (2 WMMA) ->64 (4 WMMA) ->16 (2 WMMA, K=64) ->8 (1 WMMA) ->1 (VALU)
// ---------------------------------------------------------------------------
__global__ __launch_bounds__(256) void mlp_score_kernel(
    const float* __restrict__ src,                      // (8,6,NPTS), batch 0
    const float* __restrict__ W1, const float* __restrict__ b1,
    const float* __restrict__ W2, const float* __restrict__ b2,
    const float* __restrict__ Wa, const float* __restrict__ ba,
    const float* __restrict__ Wb, const float* __restrict__ bb,
    const float* __restrict__ Wc, const float* __restrict__ bc,
    float* __restrict__ scores) {
  __shared__ __attribute__((aligned(32))) _Float16 wA[9][32][16]; // 9 KB
  __shared__ float bias_s[32 + 64 + 16 + 8];                      // staged biases
  __shared__ float act_s[8][64 * 16];                             // 32 KB

  const int tid = threadIdx.x;

  // ---- Cooperative staging: A tiles pre-swizzled to WMMA layout ----------
  // Element j of lane's 16-half vector maps to K = kwin_base(j) as below.
  auto stage = [&](int t, const float* __restrict__ W, int R, int Cin,
                   int mrow, int kbase) {
#pragma unroll
    for (int it = 0; it < 2; ++it) {                 // 2*256 = 512 elements
      const int idx   = it * 256 + tid;
      const int lane  = idx >> 4;
      const int j     = idx & 15;
      const int half_ = lane >> 4;
      const int lrow  = lane & 15;
      const int row   = mrow + lrow;
      const int v     = j >> 1;
      const int k     = kbase + half_ * 16 + 2 * v
                        + (j & 1);
      const bool valid = (row < R) && (k < Cin);
      float w = W[(valid ? row : 0) * Cin + (valid ? k : 0)]; // clamped load
      wA[t][lane][j] = (_Float16)(valid ? w : 0.f);
    }
  };
  stage(0, W1, 32, 6,  0,  0);
  stage(1, W1, 32, 6,  16, 0);
  stage(2, W2, 64, 32, 0,  0);
  stage(3, W2, 64, 32, 16, 0);
  stage(4, W2, 64, 32, 32, 0);
  stage(5, W2, 64, 32, 48, 0);
  stage(6, Wa, 16, 64, 0,  0);
  stage(7, Wa, 16, 64, 0,  32);
  stage(8, Wb, 8,  16, 0,  0);
  if (tid < 32)           bias_s[tid]           = b1[tid];
  if (tid >= 32 && tid < 96)  bias_s[tid]       = b2[tid - 32];
  if (tid >= 96 && tid < 112) bias_s[tid]       = ba[tid - 96];
  if (tid >= 112 && tid < 120) bias_s[tid]      = bb[tid - 112];
  __syncthreads();

  const float* bs1 = bias_s;
  const float* bs2 = bias_s + 32;
  const float* bsa = bias_s + 96;
  const float* bsb = bias_s + 112;

  const int lane   = tid & 31;
  const int waveId = tid >> 5;
  const int tile   = blockIdx.x * 8 + waveId;
  const int n0     = tile * 16;
  float* act = act_s[waveId];
  const int half_ = lane >> 4;
  const int lcol  = lane & 15;
  v8f zero = {};

  auto loadA = [&](int t) {
    return *reinterpret_cast<const v16h*>(&wA[t][lane][0]); // 2x ds_load_b128
  };

  // ---- Layer 1: 6 -> 32. Unconditional channel loads, mask via select. ----
  float x[6];
#pragma unroll
  for (int c = 0; c < 6; ++c) x[c] = src[c * NPTS + n0 + lcol];
  const float msk = (half_ == 0) ? 1.f : 0.f;   // lanes 16-31 carry K=16..31=0
  v16h b0;
#pragma unroll
  for (int j = 0; j < 16; ++j)
    b0[j] = (j < 6) ? (_Float16)(x[j < 6 ? j : 0] * msk) : (_Float16)0.f;

  v8f h0 = wmma16(loadA(0), b0, zero);
  v8f h1 = wmma16(loadA(1), b0, zero);
  store_D_relu(act, h0, bs1, 0,  32, lane);
  store_D_relu(act, h1, bs1, 16, 32, lane);
  __syncthreads();

  // ---- Layer 2: 32 -> 64 ----
  v16h bL2 = load_B_lds(act, 0, 32, lane);
  v8f g0 = wmma16(loadA(2), bL2, zero);
  v8f g1 = wmma16(loadA(3), bL2, zero);
  v8f g2 = wmma16(loadA(4), bL2, zero);
  v8f g3 = wmma16(loadA(5), bL2, zero);
  __syncthreads();
  store_D_relu(act, g0, bs2, 0,  64, lane);
  store_D_relu(act, g1, bs2, 16, 64, lane);
  store_D_relu(act, g2, bs2, 32, 64, lane);
  store_D_relu(act, g3, bs2, 48, 64, lane);
  __syncthreads();

  // ---- Layer a: 64 -> 16 (two accumulating WMMAs over K) ----
  v16h bA0 = load_B_lds(act, 0,  64, lane);
  v16h bA1 = load_B_lds(act, 32, 64, lane);
  v8f a0 = wmma16(loadA(6), bA0, zero);
  a0     = wmma16(loadA(7), bA1, a0);
  __syncthreads();
  store_D_relu(act, a0, bsa, 0, 16, lane);
  __syncthreads();

  // ---- Layer b: 16 -> 8 (K padded to 32) ----
  v16h bB0 = load_B_lds(act, 0, 16, lane);
  v8f c0 = wmma16(loadA(8), bB0, zero);
  __syncthreads();
  store_D_relu(act, c0, bsb, 0, 8, lane);
  __syncthreads();

  // ---- Layer c: 8 -> 1, softplus; lanes 0-15 emit one score each ----
  if (half_ == 0) {
    float s = bc[0];
#pragma unroll
    for (int k = 0; k < 8; ++k) s += Wc[k] * act[k * 16 + lcol];
    // softplus(x) = max(x,0) + log1p(exp(-|x|))  (stable form)
    float sp = fmaxf(s, 0.f) + log1pf(expf(-fabsf(s)));
    scores[n0 + lcol] = sp;
  }
}

// ---------------------------------------------------------------------------
// Kernel 2: deterministic top-64 via 64 masked arg-max rounds.
// Single 1024-thread block; ties broken by smaller index (matches lax.top_k).
// ---------------------------------------------------------------------------
__global__ __launch_bounds__(1024) void topk_kernel(float* __restrict__ scores,
                                                    int* __restrict__ topk) {
  __shared__ float sval[1024];
  __shared__ int   sidx[1024];
  const int tid = threadIdx.x;
  for (int iter = 0; iter < KKEY; ++iter) {
    float bv = -INFINITY;
    int   bi = 0x7fffffff;
    for (int i = tid; i < NPTS; i += 1024) {
      float v = scores[i];
      if (v > bv || (v == bv && i < bi)) { bv = v; bi = i; }
    }
    sval[tid] = bv;
    sidx[tid] = bi;
    __syncthreads();
    for (int off = 512; off > 0; off >>= 1) {
      if (tid < off) {
        float v2 = sval[tid + off];
        int   i2 = sidx[tid + off];
        if (v2 > sval[tid] || (v2 == sval[tid] && i2 < sidx[tid])) {
          sval[tid] = v2;
          sidx[tid] = i2;
        }
      }
      __syncthreads();
    }
    if (tid == 0) {
      topk[iter] = sidx[0];
      scores[sidx[0]] = -INFINITY;   // mask for next round
    }
    __syncthreads();
  }
}

// ---------------------------------------------------------------------------
// Kernel 3: out[b,k,c] = src[b,c,idx[k]]  -> (8,64,6) row-major
// ---------------------------------------------------------------------------
__global__ void gather_kernel(const float* __restrict__ src,
                              const int* __restrict__ topk,
                              float* __restrict__ out) {
  int t = blockIdx.x * blockDim.x + threadIdx.x;
  if (t >= 8 * KKEY * 6) return;
  int c = t % 6;
  int k = (t / 6) % KKEY;
  int b = t / (6 * KKEY);
  out[t] = src[(b * 6 + c) * NPTS + topk[k]];
}

// ---------------------------------------------------------------------------
extern "C" void kernel_launch(void* const* d_in, const int* in_sizes, int n_in,
                              void* d_out, int out_size, void* d_ws, size_t ws_size,
                              hipStream_t stream) {
  const float* src = (const float*)d_in[0];   // src_pts (8,6,65536)
  // d_in[1] = tgt_pts (unused by the reference's return value)
  const float* W1 = (const float*)d_in[2];
  const float* b1 = (const float*)d_in[3];
  const float* W2 = (const float*)d_in[4];
  const float* b2 = (const float*)d_in[5];
  const float* Wa = (const float*)d_in[6];
  const float* ba = (const float*)d_in[7];
  const float* Wb = (const float*)d_in[8];
  const float* bb = (const float*)d_in[9];
  const float* Wc = (const float*)d_in[10];
  const float* bc = (const float*)d_in[11];

  float* scores = (float*)d_ws;                                   // 65536 f32
  int*   topk   = (int*)((char*)d_ws + NPTS * sizeof(float));     // 64 i32
  float* out    = (float*)d_out;                                  // (8,64,6)

  // 4096 16-point tiles, 8 waves (tiles) per 256-thread block -> 512 blocks.
  mlp_score_kernel<<<NPTS / (16 * 8), 256, 0, stream>>>(
      src, W1, b1, W2, b2, Wa, ba, Wb, bb, Wc, bc, scores);
  topk_kernel<<<1, 1024, 0, stream>>>(scores, topk);
  gather_kernel<<<(8 * KKEY * 6 + 255) / 256, 256, 0, stream>>>(src, topk, out);
}
